// LRUModel_77068893160204
// MI455X (gfx1250) — compile-verified, matrix operand order checked by reading, not run on hardware
//
#include <hip/hip_runtime.h>

typedef __attribute__((ext_vector_type(16))) _Float16 v16h;
typedef __attribute__((ext_vector_type(8)))  float    v8f;
typedef __attribute__((ext_vector_type(4)))  int      v4i;

#define HID 64
#define SEQ 24
#define MEMS 8
#define NTOK 66             // VOCAB_SIZE + 2 embedding rows
#define EWSTRIDE 132        // padded f32 row stride for EW in LDS (132 mod 64 dwords = 4)
#define W2STRIDE 72         // padded f16 row stride (36 dwords -> 16 distinct banks)
#define HSTRIDE  72
#define ROWS_PER_BLOCK 64   // 4 waves * 16 rows

// ---------------------------------------------------------------------------
// Pre-kernel: EW[t][0:64]   = embed_W[t] @ W1[:, :64]^T  + b1   (query half)
//             EW[t][64:128] = embed_W[t] @ W1[:, 64:]^T         (memory half)
// 66*128 outputs, 64-long dots; runs once, feeds every block of the main pass.
// ---------------------------------------------------------------------------
__global__ void __launch_bounds__(256)
precompute_ew(const float* __restrict__ embed_W,
              const float* __restrict__ W1,
              const float* __restrict__ b1,
              float* __restrict__ EWg) {
    const int idx = blockIdx.x * 256 + threadIdx.x;
    if (idx >= NTOK * 2 * HID) return;
    const int t    = idx >> 7;
    const int cc   = idx & 127;
    const int half = cc >> 6;
    const int c    = cc & 63;
    const float* e = embed_W + t * HID;
    const float* w = W1 + c * (2 * HID) + half * HID;
    float s = 0.f;
#pragma unroll 8
    for (int k = 0; k < HID; ++k) s += e[k] * w[k];
    if (half == 0) s += b1[c];
    EWg[idx] = s;
}

// Fragment builder for v_wmma_f32_16x16x32_f16 (16-bit A/B layout):
// lanes 0-15: K = kBase+[0..7] (elems 0..7), kBase+16..23 (elems 8..15);
// lanes 16-31 shift K by +8. kh = lane>>4. Two contiguous 16B LDS reads.
__device__ __forceinline__ v16h frag16(const _Float16* __restrict__ row, int kBase, int kh) {
    const _Float16* p = row + kBase + kh * 8;
    v16h f;
#pragma unroll
    for (int i = 0; i < 8; ++i) {
        f[i]     = p[i];
        f[i + 8] = p[16 + i];
    }
    return f;
}

__device__ __forceinline__ v8f wmma16(v16h a, v16h b, v8f c) {
    return __builtin_amdgcn_wmma_f32_16x16x32_f16(
        false, a, false, b, (short)0, c, false, false);
}

__global__ void __launch_bounds__(128)
lru_fused_kernel(const int* __restrict__ seqs,
                 const int* __restrict__ query_tok,
                 const float* __restrict__ W2,
                 const float* __restrict__ b2,
                 const float* __restrict__ EWg,
                 float* __restrict__ out) {
    __shared__ float    EW[NTOK * EWSTRIDE];         // 34848 B
    __shared__ _Float16 W2F[HID * W2STRIDE];         //  9216 B
    __shared__ float    B2[HID];                     //   256 B
    __shared__ _Float16 H[ROWS_PER_BLOCK * HSTRIDE]; //  9216 B

    const int tid  = threadIdx.x;
    const int row0 = blockIdx.x * ROWS_PER_BLOCK;

    // ---- Stage precomputed EW (f32, padded) and W2 (f16, padded) ----
    // Regular-temporal loads: shared by all 4096 blocks, should stay in L2.
    for (int i = tid; i < NTOK * 2 * HID; i += 128) {
        EW[(i >> 7) * EWSTRIDE + (i & 127)] = EWg[i];
    }
    for (int i = tid; i < HID * HID; i += 128) {
        W2F[(i >> 6) * W2STRIDE + (i & 63)] = (_Float16)W2[i];
    }
    if (tid < HID) B2[tid] = b2[tid];
    __syncthreads();

    // ---- Phase 1: h = relu(EW[q][:64] + 0.125 * sum_j EW[tok_j][64:]) ----
    // 2 threads per row (32 cols each); wave w produces rows 16w..16w+15.
    {
        const int rb = tid >> 1;
        const int ch = tid & 1;
        const int r  = row0 + rb;
        const int q  = __builtin_nontemporal_load(query_tok + r);
        // tokens 15..22: three aligned 16B non-temporal loads (cols 12..23)
        const v4i* sp = (const v4i*)(seqs + r * SEQ + 12);
        const v4i t0 = __builtin_nontemporal_load(sp + 0);
        const v4i t1 = __builtin_nontemporal_load(sp + 1);
        const v4i t2 = __builtin_nontemporal_load(sp + 2);
        const int toks[MEMS] = { t0[3], t1[0], t1[1], t1[2], t1[3], t2[0], t2[1], t2[2] };

        float sm[32];
#pragma unroll
        for (int c = 0; c < 32; ++c) sm[c] = 0.f;
#pragma unroll
        for (int j = 0; j < MEMS; ++j) {
            const float* em = &EW[toks[j] * EWSTRIDE + HID + ch * 32];
#pragma unroll
            for (int c = 0; c < 32; ++c) sm[c] += em[c];
        }
        const float* eq = &EW[q * EWSTRIDE + ch * 32];
        _Float16* dst = &H[rb * HSTRIDE + ch * 32];
#pragma unroll
        for (int c = 0; c < 32; ++c) {
            float h = eq[c] + 0.125f * sm[c];   // b1 already folded into eq
            h = h > 0.f ? h : 0.f;
            dst[c] = (_Float16)h;
        }
    }
    __syncthreads();

    // ---- Phase 2: logits = h @ W2^T + b2 via WMMA (16 rows per wave) ----
    const int wv   = tid >> 5;
    const int lane = tid & 31;
    const int m    = lane & 15;
    const int kh   = lane >> 4;

    const _Float16* aRow = &H[(wv * 16 + m) * HSTRIDE];
    const v16h h0 = frag16(aRow,  0, kh);
    const v16h h1 = frag16(aRow, 32, kh);

    float* outBase = out + (size_t)(row0 + wv * 16) * HID;
#pragma unroll
    for (int nt = 0; nt < 4; ++nt) {
        const _Float16* brow = &W2F[(nt * 16 + m) * W2STRIDE];
        v8f c = {};
        c = wmma16(h0, frag16(brow,  0, kh), c);
        c = wmma16(h1, frag16(brow, 32, kh), c);
        const float bv = B2[nt * 16 + m];
#pragma unroll
        for (int v = 0; v < 8; ++v) {
            const int M = v + kh * 8;   // C layout: VGPR v -> row v + 8*(lane>>4)
            // Touch-once 64MB output stream: non-temporal store keeps L2 clean.
            __builtin_nontemporal_store(c[v] + bv, outBase + M * HID + nt * 16 + m);
        }
    }
}

extern "C" void kernel_launch(void* const* d_in, const int* in_sizes, int n_in,
                              void* d_out, int out_size, void* d_ws, size_t ws_size,
                              hipStream_t stream) {
    const int*   seqs  = (const int*)  d_in[0];
    const int*   qtok  = (const int*)  d_in[1];
    const float* embed = (const float*)d_in[2];
    const float* W1    = (const float*)d_in[3];
    const float* b1    = (const float*)d_in[4];
    const float* W2    = (const float*)d_in[5];
    const float* b2    = (const float*)d_in[6];
    float* out = (float*)d_out;
    float* EWg = (float*)d_ws;                  // 66*128 f32 = 33792 B scratch

    // Fold embed_W @ W1^T (+b1) once; feeds every block of the main kernel.
    const int ew_elems = NTOK * 2 * HID;
    precompute_ew<<<(ew_elems + 255) / 256, 256, 0, stream>>>(embed, W1, b1, EWg);

    const int B = in_sizes[1];                  // query_tok element count
    lru_fused_kernel<<<B / ROWS_PER_BLOCK, 128, 0, stream>>>(seqs, qtok, W2, b2, EWg, out);
}
